// RPN_81037442941095
// MI455X (gfx1250) — matrix operand order repaired and checked by reading, not compiled, not run on hardware
//
#include <hip/hip_runtime.h>
#include <hip/hip_bf16.h>
#include <math.h>

typedef _Float16 half8  __attribute__((ext_vector_type(8)));
typedef _Float16 half16 __attribute__((ext_vector_type(16)));
typedef float    float8 __attribute__((ext_vector_type(8)));

#define B_    4
#define C_    512
#define H_    64
#define W_    96
#define HW_   (H_ * W_)          // 6144
#define M_    (B_ * HW_)         // 24576
#define A_    9
#define NSC   (B_ * A_ * HW_)    // 221184
#define KSEL  2048
#define NOUT  50
#define THR_  0.05f
#define IOU_  0.7f
#define NBIN  4096

// ---- workspace byte offsets ----
static constexpr size_t OFF_XH   = 0;                                   // f16 NHWC input  [M_][C_]
static constexpr size_t OFF_WP   = OFF_XH  + (size_t)M_ * C_ * 2;       // f16 w1 packed   [9][C_][C_] (n-major,c contig)
static constexpr size_t OFF_WCR  = OFF_WP  + (size_t)9 * C_ * C_ * 2;   // f16 cls+reg w   [64][C_]
static constexpr size_t OFF_B64  = OFF_WCR + (size_t)64 * C_ * 2;       // f32 bias        [64]
static constexpr size_t OFF_FH   = OFF_B64 + 256;                       // f16 conv3 out   [M_][C_]
static constexpr size_t OFF_O54  = OFF_FH  + (size_t)M_ * C_ * 2;       // f32 cls+reg out [M_][64]
static constexpr size_t OFF_SC   = OFF_O54 + (size_t)M_ * 64 * 4;       // f32 scores      [NSC]
static constexpr size_t OFF_HIST = OFF_SC  + (size_t)NSC * 4;           // u32 hist        [NBIN]
static constexpr size_t OFF_HDR  = OFF_HIST + (size_t)NBIN * 4;         // i32 header      [16]
static constexpr size_t OFF_TS   = OFF_HDR + 64;                        // f32 top scores  [KSEL]
static constexpr size_t OFF_TI   = OFF_TS  + (size_t)KSEL * 4;          // i32 top idx     [KSEL]
static constexpr size_t OFF_BOX  = OFF_TI  + (size_t)KSEL * 4;          // f32 boxes       [KSEL][4]
static constexpr size_t OFF_AREA = OFF_BOX + (size_t)KSEL * 16;         // f32 area        [KSEL]
static constexpr size_t OFF_VAL  = OFF_AREA + (size_t)KSEL * 4;         // i32 valid       [KSEL]

__constant__ float ANCW[9] = {128.f,128.f,256.f, 256.f,256.f,512.f, 512.f,512.f,1024.f};
__constant__ float ANCH[9] = {128.f,256.f,128.f, 256.f,512.f,256.f, 512.f,1024.f,512.f};

// ---------------- packing kernels ----------------
__global__ void pack_x(const float* __restrict__ x, _Float16* __restrict__ xh) {
  int o = blockIdx.x * blockDim.x + threadIdx.x;
  if (o >= M_ * C_) return;
  int c  = o % C_;
  int px = (o / C_) % W_;
  int y  = (o / (C_ * W_)) % H_;
  int b  = o / (C_ * W_ * H_);
  xh[o] = (_Float16)x[(((size_t)b * C_ + c) * H_ + y) * W_ + px];
}

__global__ void pack_w(const float* __restrict__ w1, _Float16* __restrict__ wp) {
  int o = blockIdx.x * blockDim.x + threadIdx.x;
  if (o >= 9 * C_ * C_) return;
  int ci  = o % C_;
  int oc  = (o / C_) % C_;
  int kyx = o / (C_ * C_);
  int ky = kyx / 3, kx = kyx % 3;
  wp[o] = (_Float16)w1[(((size_t)oc * C_ + ci) * 3 + ky) * 3 + kx];
}

__global__ void pack_wcr(const float* __restrict__ wcls, const float* __restrict__ bcls,
                         const float* __restrict__ wreg, const float* __restrict__ breg,
                         _Float16* __restrict__ wcr, float* __restrict__ b64) {
  int o = blockIdx.x * blockDim.x + threadIdx.x;
  if (o >= 64 * C_) return;
  int c = o % C_;
  int n = o / C_;
  float v = 0.0f;
  if (n < 18)       v = wcls[(size_t)n * C_ + c];
  else if (n < 54)  v = wreg[(size_t)(n - 18) * C_ + c];
  wcr[o] = (_Float16)v;
  if (c == 0) b64[n] = (n < 18) ? bcls[n] : (n < 54 ? breg[n - 18] : 0.0f);
}

// ---------------- 3x3 conv as implicit GEMM (WMMA f16 -> f32 acc) ----------------
// M = 24576 pixels, N = 512, K = 9*512. One wave: 32(M) x 64(N) tile.
// Per K-step: 2 A-frags + 4 B-frags -> 8 WMMAs (1.5 b128 loads per WMMA).
__global__ __launch_bounds__(256)
void conv3_wmma(const _Float16* __restrict__ xh, const _Float16* __restrict__ wp,
                const float* __restrict__ b1, _Float16* __restrict__ fh) {
  int lane  = threadIdx.x & 31;
  int gw    = (blockIdx.x * (blockDim.x >> 5)) + (threadIdx.x >> 5);
  int nGrp  = gw & 7;       // 8 groups of 64 out-channels
  int mTile = gw >> 3;      // 768 tiles of 32 pixels (rows never split: 96 = 3*32)
  if (mTile >= M_ / 32) return;

  int x0 = (mTile % (W_ / 32)) * 32;
  int y  = (mTile / (W_ / 32)) % H_;
  int b  = mTile / ((W_ / 32) * H_);
  int mrow  = lane & 15;
  int khalf = lane >> 4;
  int ko8   = khalf * 8;
  int kb16  = khalf * 16;
  int nl    = lane & 15;
  int n0    = nGrp * 64;

  float8 acc00 = {}, acc01 = {}, acc02 = {}, acc03 = {};
  float8 acc10 = {}, acc11 = {}, acc12 = {}, acc13 = {};

  for (int kyx = 0; kyx < 9; ++kyx) {
    int ky = kyx / 3, kx = kyx % 3;
    int yy  = y + ky - 1;
    int xxa = x0 + mrow + kx - 1;        // A-frag 0 (pixels x0..x0+15)
    int xxb = xxa + 16;                  // A-frag 1 (pixels x0+16..x0+31)
    bool rowok = (yy >= 0) && (yy < H_);
    bool inb0 = rowok && (xxa >= 0) && (xxa < W_);
    bool inb1 = rowok && (xxb >= 0) && (xxb < W_);
    int yyc  = min(max(yy, 0), H_ - 1);
    int xxa_c = min(max(xxa, 0), W_ - 1);
    int xxb_c = min(max(xxb, 0), W_ - 1);
    const _Float16* apix0 = xh + (size_t)((b * H_ + yyc) * W_ + xxa_c) * C_;
    const _Float16* apix1 = xh + (size_t)((b * H_ + yyc) * W_ + xxb_c) * C_;
    const _Float16* bbase = wp + (size_t)kyx * C_ * C_;

    for (int c0 = 0; c0 < C_; c0 += 32) {
      half16 a0, a1;
      if (inb0) {
        a0 = *(const half16*)(apix0 + c0 + kb16);
      } else {
#pragma unroll
        for (int i = 0; i < 16; ++i) a0[i] = (_Float16)0.0f;
      }
      if (inb1) {
        a1 = *(const half16*)(apix1 + c0 + kb16);
      } else {
#pragma unroll
        for (int i = 0; i < 16; ++i) a1[i] = (_Float16)0.0f;
      }
      const _Float16* bp = bbase + (size_t)(n0 + nl) * C_ + c0 + kb16;
      half16 bf0 = *(const half16*)(bp);
      half16 bf1 = *(const half16*)(bp + 16 * C_);
      half16 bf2 = *(const half16*)(bp + 32 * C_);
      half16 bf3 = *(const half16*)(bp + 48 * C_);
      acc00 = __builtin_amdgcn_wmma_f32_16x16x32_f16(false, a0, false, bf0, (short)0, acc00, false, false);
      acc01 = __builtin_amdgcn_wmma_f32_16x16x32_f16(false, a0, false, bf1, (short)0, acc01, false, false);
      acc02 = __builtin_amdgcn_wmma_f32_16x16x32_f16(false, a0, false, bf2, (short)0, acc02, false, false);
      acc03 = __builtin_amdgcn_wmma_f32_16x16x32_f16(false, a0, false, bf3, (short)0, acc03, false, false);
      acc10 = __builtin_amdgcn_wmma_f32_16x16x32_f16(false, a1, false, bf0, (short)0, acc10, false, false);
      acc11 = __builtin_amdgcn_wmma_f32_16x16x32_f16(false, a1, false, bf1, (short)0, acc11, false, false);
      acc12 = __builtin_amdgcn_wmma_f32_16x16x32_f16(false, a1, false, bf2, (short)0, acc12, false, false);
      acc13 = __builtin_amdgcn_wmma_f32_16x16x32_f16(false, a1, false, bf3, (short)0, acc13, false, false);
    }
  }

  int pbase = (b * H_ + y) * W_;
  float bb0 = b1[n0 + nl], bb1 = b1[n0 + 16 + nl], bb2 = b1[n0 + 32 + nl], bb3 = b1[n0 + 48 + nl];
#pragma unroll
  for (int v = 0; v < 8; ++v) {
    int px0 = x0 + v + 8 * khalf;
    _Float16* dst0 = fh + (size_t)(pbase + px0) * C_ + n0 + nl;
    dst0[0]  = (_Float16)(acc00[v] + bb0);
    dst0[16] = (_Float16)(acc01[v] + bb1);
    dst0[32] = (_Float16)(acc02[v] + bb2);
    dst0[48] = (_Float16)(acc03[v] + bb3);
    _Float16* dst1 = fh + (size_t)(pbase + px0 + 16) * C_ + n0 + nl;
    dst1[0]  = (_Float16)(acc10[v] + bb0);
    dst1[16] = (_Float16)(acc11[v] + bb1);
    dst1[32] = (_Float16)(acc12[v] + bb2);
    dst1[48] = (_Float16)(acc13[v] + bb3);
  }
}

// ---------------- fused 1x1 cls(18)+reg(36) conv, N padded to 64 ----------------
__global__ __launch_bounds__(256)
void conv1_wmma(const _Float16* __restrict__ fh, const _Float16* __restrict__ wcr,
                const float* __restrict__ b64, float* __restrict__ out54) {
  int lane = threadIdx.x & 31;
  int gw   = (blockIdx.x * (blockDim.x >> 5)) + (threadIdx.x >> 5);
  if (gw >= M_ / 16) return;
  int m0    = gw * 16;
  int mrow  = lane & 15;
  int khalf = lane >> 4;
  int ko8   = khalf * 8;
  int kb16  = khalf * 16;
  int nl    = lane & 15;

  const _Float16* arow = fh + (size_t)(m0 + mrow) * C_;
  float8 acc0 = {}, acc1 = {}, acc2 = {}, acc3 = {};
  for (int c0 = 0; c0 < C_; c0 += 32) {
    half16 a;
    a = *(const half16*)(arow + c0 + kb16);
    const _Float16* bp = wcr + (size_t)nl * C_ + c0 + kb16;
    half16 bf0 = *(const half16*)(bp);
    half16 bf1 = *(const half16*)(bp + 16 * C_);
    half16 bf2 = *(const half16*)(bp + 32 * C_);
    half16 bf3 = *(const half16*)(bp + 48 * C_);
    acc0 = __builtin_amdgcn_wmma_f32_16x16x32_f16(false, a, false, bf0, (short)0, acc0, false, false);
    acc1 = __builtin_amdgcn_wmma_f32_16x16x32_f16(false, a, false, bf1, (short)0, acc1, false, false);
    acc2 = __builtin_amdgcn_wmma_f32_16x16x32_f16(false, a, false, bf2, (short)0, acc2, false, false);
    acc3 = __builtin_amdgcn_wmma_f32_16x16x32_f16(false, a, false, bf3, (short)0, acc3, false, false);
  }
  float bb0 = b64[nl], bb1 = b64[16 + nl], bb2 = b64[32 + nl], bb3 = b64[48 + nl];
#pragma unroll
  for (int v = 0; v < 8; ++v) {
    int m = m0 + v + 8 * khalf;
    float* dst = out54 + (size_t)m * 64 + nl;
    dst[0]  = acc0[v] + bb0;
    dst[16] = acc1[v] + bb1;
    dst[32] = acc2[v] + bb2;
    dst[48] = acc3[v] + bb3;
  }
}

// ---------------- softmax over 18 channels -> fg scores [B,A,H,W] ----------------
__global__ void softmax_scores(const float* __restrict__ out54, float* __restrict__ scores) {
  int p = blockIdx.x * blockDim.x + threadIdx.x;
  if (p >= M_) return;
  const float* row = out54 + (size_t)p * 64;
  float mx = row[0];
#pragma unroll
  for (int i = 1; i < 18; ++i) mx = fmaxf(mx, row[i]);
  float e[18]; float sum = 0.0f;
#pragma unroll
  for (int i = 0; i < 18; ++i) { e[i] = expf(row[i] - mx); sum += e[i]; }
  float inv = 1.0f / sum;
  int b = p / HW_, q = p % HW_;
  float* sb = scores + (size_t)b * A_ * HW_ + q;
#pragma unroll
  for (int a = 0; a < A_; ++a) sb[(size_t)a * HW_] = e[2 * a + 1] * inv;
}

// ---------------- top-K selection: histogram -> cutoff -> collect -> bitonic ----------------
__global__ void init_sel(unsigned int* __restrict__ hist, float* __restrict__ top_s, int* __restrict__ top_i) {
  int i = blockIdx.x * blockDim.x + threadIdx.x;
  if (i < NBIN) hist[i] = 0u;
  if (i < KSEL) { top_s[i] = -1.0f; top_i[i] = 0; }
}

__global__ void score_hist(const float* __restrict__ scores, unsigned int* __restrict__ hist) {
  int i = blockIdx.x * blockDim.x + threadIdx.x;
  if (i >= NSC) return;
  float s = scores[i];
  if (s >= THR_) atomicAdd(&hist[__float_as_uint(s) >> 20], 1u);
}

__global__ void find_cutoff(const unsigned int* __restrict__ hist, int* __restrict__ hdr) {
  if (threadIdx.x != 0 || blockIdx.x != 0) return;
  int acc = 0, cut = -1;
  for (int b = NBIN - 1; b >= 0; --b) {
    int h = (int)hist[b];
    if (acc + h >= KSEL) { cut = b; break; }
    acc += h;
  }
  hdr[0] = cut; hdr[1] = acc; hdr[2] = 0; hdr[3] = 0;
}

__global__ void collect_topk(const float* __restrict__ scores, int* __restrict__ hdr,
                             float* __restrict__ top_s, int* __restrict__ top_i) {
  int i = blockIdx.x * blockDim.x + threadIdx.x;
  if (i >= NSC) return;
  float s = scores[i];
  if (s < THR_) return;
  int bin = (int)(__float_as_uint(s) >> 20);
  int cut = hdr[0], cab = hdr[1];
  if (bin > cut) {
    int pos = (int)atomicAdd((unsigned int*)&hdr[2], 1u);
    if (pos < KSEL) { top_s[pos] = s; top_i[pos] = i; }
  } else if (bin == cut) {
    int pos = (int)atomicAdd((unsigned int*)&hdr[3], 1u);
    if (pos < KSEL - cab) { top_s[cab + pos] = s; top_i[cab + pos] = i; }
  }
}

__global__ __launch_bounds__(1024)
void bitonic_sort(float* __restrict__ top_s, int* __restrict__ top_i) {
  __shared__ float ss[KSEL];
  __shared__ int   si[KSEL];
  int tid = threadIdx.x;
  ss[tid] = top_s[tid];               si[tid] = top_i[tid];
  ss[tid + 1024] = top_s[tid + 1024]; si[tid + 1024] = top_i[tid + 1024];
  for (int k = 2; k <= KSEL; k <<= 1) {
    for (int j = k >> 1; j > 0; j >>= 1) {
      __syncthreads();
#pragma unroll
      for (int s2 = 0; s2 < 2; ++s2) {
        int i = tid + s2 * 1024;
        int ixj = i ^ j;
        if (ixj > i) {
          float a = ss[i], b = ss[ixj];
          bool descSeg = ((i & k) == 0);
          bool sw = descSeg ? (a < b) : (a > b);
          if (sw) { ss[i] = b; ss[ixj] = a; int t = si[i]; si[i] = si[ixj]; si[ixj] = t; }
        }
      }
    }
  }
  __syncthreads();
  top_s[tid] = ss[tid];               top_i[tid] = si[tid];
  top_s[tid + 1024] = ss[tid + 1024]; top_i[tid + 1024] = si[tid + 1024];
}

// ---------------- box decode ----------------
__global__ void decode_boxes(const float* __restrict__ top_s, const int* __restrict__ top_i,
                             const float* __restrict__ out54, const int* __restrict__ hp,
                             const int* __restrict__ wpv, float* __restrict__ boxes,
                             float* __restrict__ area, int* __restrict__ validf) {
  int k = blockIdx.x * blockDim.x + threadIdx.x;
  if (k >= KSEL) return;
  float s = top_s[k];
  if (s >= THR_) {
    int i = top_i[k];
    int b   = i / (A_ * HW_);
    int rem = i % (A_ * HW_);
    int a   = rem / HW_;
    int q   = rem % HW_;
    int yi = q / W_, xi = q % W_;
    float offx = (float)((wpv[0] - 16) % 16) * 0.5f + 8.0f;
    float offy = (float)((hp[0]  - 16) % 16) * 0.5f + 8.0f;
    float cx = (float)xi * 16.0f + offx;
    float cy = (float)yi * 16.0f + offy;
    float aw = ANCW[a], ah = ANCH[a];
    const float* r = out54 + (size_t)(b * HW_ + q) * 64 + 18 + a * 4;
    float bx = r[0] * aw + cx;
    float by = r[1] * ah + cy;
    float bw = expf(r[2]) * aw;
    float bh = expf(r[3]) * ah;
    float x1 = bx - bw * 0.5f, y1 = by - bh * 0.5f;
    float x2 = bx + bw * 0.5f, y2 = by + bh * 0.5f;
    boxes[k * 4 + 0] = x1; boxes[k * 4 + 1] = y1;
    boxes[k * 4 + 2] = x2; boxes[k * 4 + 3] = y2;
    area[k] = (x2 - x1) * (y2 - y1);
    validf[k] = 1;
  } else {
    boxes[k * 4 + 0] = 0.f; boxes[k * 4 + 1] = 0.f;
    boxes[k * 4 + 2] = 0.f; boxes[k * 4 + 3] = 0.f;
    area[k] = 0.f; validf[k] = 0;
  }
}

// ---------------- greedy NMS (matches reference sequential suppression) + final top-50 ----------------
__global__ __launch_bounds__(1024)
void nms_final(const float* __restrict__ top_s, const float* __restrict__ boxes,
               const float* __restrict__ area, const int* __restrict__ validf,
               float* __restrict__ out) {
  __shared__ float sx1[KSEL], sy1[KSEL], sx2[KSEL], sy2[KSEL], sar[KSEL];
  __shared__ unsigned char keep[KSEL];
  int tid = threadIdx.x;
  for (int k = tid; k < KSEL; k += 1024) {
    sx1[k] = boxes[k * 4 + 0]; sy1[k] = boxes[k * 4 + 1];
    sx2[k] = boxes[k * 4 + 2]; sy2[k] = boxes[k * 4 + 3];
    sar[k] = area[k];
    keep[k] = (unsigned char)(validf[k] != 0);
  }
  __syncthreads();
  for (int i = 0; i < KSEL - 1; ++i) {
    if (keep[i]) {
      float ax1 = sx1[i], ay1 = sy1[i], ax2 = sx2[i], ay2 = sy2[i], aar = sar[i];
      for (int j = i + 1 + tid; j < KSEL; j += 1024) {
        if (keep[j]) {
          float iw = fminf(ax2, sx2[j]) - fmaxf(ax1, sx1[j]);
          float ih = fminf(ay2, sy2[j]) - fmaxf(ay1, sy1[j]);
          float inter = fmaxf(iw, 0.f) * fmaxf(ih, 0.f);
          float iou = inter / (aar + sar[j] - inter + 1e-9f);
          if (iou > IOU_) keep[j] = 0;
        }
      }
    }
    __syncthreads();
  }
  if (tid == 0) {
    int cnt = 0;
    for (int k = 0; k < KSEL && cnt < NOUT; ++k) {
      if (keep[k]) {
        out[cnt] = top_s[k];
        out[NOUT + cnt * 4 + 0] = sx1[k];
        out[NOUT + cnt * 4 + 1] = sy1[k];
        out[NOUT + cnt * 4 + 2] = sx2[k];
        out[NOUT + cnt * 4 + 3] = sy2[k];
        ++cnt;
      }
    }
    for (; cnt < NOUT; ++cnt) {
      out[cnt] = 0.f;
      out[NOUT + cnt * 4 + 0] = 0.f; out[NOUT + cnt * 4 + 1] = 0.f;
      out[NOUT + cnt * 4 + 2] = 0.f; out[NOUT + cnt * 4 + 3] = 0.f;
    }
  }
}

extern "C" void kernel_launch(void* const* d_in, const int* in_sizes, int n_in,
                              void* d_out, int out_size, void* d_ws, size_t ws_size,
                              hipStream_t stream) {
  (void)in_sizes; (void)n_in; (void)out_size; (void)ws_size;
  const float* x    = (const float*)d_in[0];
  const float* w1   = (const float*)d_in[1];
  const float* b1   = (const float*)d_in[2];
  const float* wcls = (const float*)d_in[3];
  const float* bcls = (const float*)d_in[4];
  const float* wreg = (const float*)d_in[5];
  const float* breg = (const float*)d_in[6];
  const int*   hp   = (const int*)d_in[7];
  const int*   wpv  = (const int*)d_in[8];
  float* out = (float*)d_out;

  char* ws = (char*)d_ws;
  _Float16* xh   = (_Float16*)(ws + OFF_XH);
  _Float16* wp   = (_Float16*)(ws + OFF_WP);
  _Float16* wcr  = (_Float16*)(ws + OFF_WCR);
  float*    b64  = (float*)(ws + OFF_B64);
  _Float16* fh   = (_Float16*)(ws + OFF_FH);
  float*    o54  = (float*)(ws + OFF_O54);
  float*    sc   = (float*)(ws + OFF_SC);
  unsigned int* hist = (unsigned int*)(ws + OFF_HIST);
  int*      hdr  = (int*)(ws + OFF_HDR);
  float*    tps  = (float*)(ws + OFF_TS);
  int*      tpi  = (int*)(ws + OFF_TI);
  float*    boxb = (float*)(ws + OFF_BOX);
  float*    areab= (float*)(ws + OFF_AREA);
  int*      valb = (int*)(ws + OFF_VAL);

  pack_x<<<(M_ * C_ + 255) / 256, 256, 0, stream>>>(x, xh);
  pack_w<<<(9 * C_ * C_ + 255) / 256, 256, 0, stream>>>(w1, wp);
  pack_wcr<<<(64 * C_ + 255) / 256, 256, 0, stream>>>(wcls, bcls, wreg, breg, wcr, b64);
  init_sel<<<(NBIN + 255) / 256, 256, 0, stream>>>(hist, tps, tpi);

  // 6144 waves (768 M-tiles of 32 pixels x 8 N-groups), 8 waves/block
  conv3_wmma<<<768, 256, 0, stream>>>(xh, wp, b1, fh);
  // 1536 waves, 8 waves/block
  conv1_wmma<<<192, 256, 0, stream>>>(fh, wcr, b64, o54);

  softmax_scores<<<(M_ + 255) / 256, 256, 0, stream>>>(o54, sc);
  score_hist<<<(NSC + 255) / 256, 256, 0, stream>>>(sc, hist);
  find_cutoff<<<1, 1, 0, stream>>>(hist, hdr);
  collect_topk<<<(NSC + 255) / 256, 256, 0, stream>>>(sc, hdr, tps, tpi);
  bitonic_sort<<<1, 1024, 0, stream>>>(tps, tpi);
  decode_boxes<<<(KSEL + 255) / 256, 256, 0, stream>>>(tps, tpi, o54, hp, wpv, boxb, areab, valb);
  nms_final<<<1, 1024, 0, stream>>>(tps, boxb, areab, valb, out);
}
